// Emu3VisionVQAttnBlock_10307921510584
// MI455X (gfx1250) — matrix operand order repaired and checked by reading, not compiled, not run on hardware
//
#include <hip/hip_runtime.h>

#define BATCH 4
#define CCH   512
#define NSP   4096
#define NGRP  32

typedef __bf16 bf16_t;
typedef __attribute__((ext_vector_type(8)))  bf16_t bf16x8;
typedef __attribute__((ext_vector_type(16))) bf16_t v16bf;
typedef __attribute__((ext_vector_type(8)))  float  v8f;

static __device__ __forceinline__ v16bf bcat(bf16x8 lo, bf16x8 hi) {
  union { v16bf v; bf16x8 h[2]; } u;
  u.h[0] = lo; u.h[1] = hi;
  return u.v;
}

static __device__ __forceinline__ v8f wmma_bf16(v16bf a, v16bf b, v8f c) {
  return __builtin_amdgcn_wmma_f32_16x16x32_bf16(false, a, false, b, (short)0, c,
                                                 false, false);
}

// ---------------------------------------------------------------------------
// GroupNorm statistics: one block per (batch, group); group data is contiguous
// (16 channels x 4096 spatial = 65536 floats).
// ---------------------------------------------------------------------------
__global__ __launch_bounds__(256) void gn_stats_kernel(const float* __restrict__ x,
                                                       float* __restrict__ mean,
                                                       float* __restrict__ rstd) {
  const int bg = blockIdx.x;
  const float* p = x + (size_t)bg * (16 * NSP);
  float s = 0.f, sq = 0.f;
  for (int e = threadIdx.x; e < 16 * NSP; e += 256) {
    float v = p[e];
    s += v;
    sq += v * v;
  }
  __shared__ float rs[256], rq[256];
  rs[threadIdx.x] = s;
  rq[threadIdx.x] = sq;
  __syncthreads();
  for (int off = 128; off > 0; off >>= 1) {
    if (threadIdx.x < off) {
      rs[threadIdx.x] += rs[threadIdx.x + off];
      rq[threadIdx.x] += rq[threadIdx.x + off];
    }
    __syncthreads();
  }
  if (threadIdx.x == 0) {
    const float inv = 1.f / (16.f * NSP);
    float m = rs[0] * inv;
    float var = rq[0] * inv - m * m;
    mean[bg] = m;
    rstd[bg] = rsqrtf(var + 1e-6f);
  }
}

// ---------------------------------------------------------------------------
// GroupNorm apply + transpose [B][C][N] f32 -> [B][N][C] bf16 via LDS tile.
// grid (N/32, C/32, B), block (32, 8)
// ---------------------------------------------------------------------------
__global__ __launch_bounds__(256) void gn_apply_kernel(
    const float* __restrict__ x, const float* __restrict__ gamma,
    const float* __restrict__ beta, const float* __restrict__ mean,
    const float* __restrict__ rstd, bf16_t* __restrict__ nxt) {
  __shared__ bf16_t tilebuf[32 * 33];
  const int b = blockIdx.z, c0 = blockIdx.y * 32, i0 = blockIdx.x * 32;
  const int tx = threadIdx.x, ty = threadIdx.y;
  for (int cc = ty; cc < 32; cc += 8) {
    const int c = c0 + cc;
    const int g = c >> 4;
    const float mu = mean[b * NGRP + g];
    const float rs = rstd[b * NGRP + g];
    const float v = x[((size_t)b * CCH + c) * NSP + i0 + tx];
    tilebuf[cc * 33 + tx] = (bf16_t)((v - mu) * rs * gamma[c] + beta[c]);
  }
  __syncthreads();
  for (int ii = ty; ii < 32; ii += 8) {
    nxt[((size_t)b * NSP + i0 + ii) * CCH + c0 + tx] = tilebuf[tx * 33 + ii];
  }
}

__global__ __launch_bounds__(256) void cvt_bf16_kernel(const float* __restrict__ s,
                                                       bf16_t* __restrict__ d, int n) {
  const int i = blockIdx.x * 256 + threadIdx.x;
  if (i < n) d[i] = (bf16_t)s[i];
}

// ---------------------------------------------------------------------------
// WMMA GEMM: D[i, c_out] = sum_c A[i, c] * W[c_out, c]  (+bias, modes below)
//   A  : bf16 [B][N][C]      W : bf16 [C][C] row-major
//   MODE 0: out_bf[B][N][C] = (D + bias) * scale        (Q / K)
//   MODE 1: out_bf[B][C][N] = D + bias  (LDS transpose) (V)
//   MODE 2: out_f [B][C][N] = resid + D + bias          (final projection)
// grid (N/64, C/64, B), 128 threads (4 waves), each wave: 16M x 64N tile
// ---------------------------------------------------------------------------
template <int MODE>
__global__ __launch_bounds__(128) void gemm_nc_kernel(
    const bf16_t* __restrict__ A, const bf16_t* __restrict__ W,
    const float* __restrict__ bias, bf16_t* __restrict__ out_bf,
    float* __restrict__ out_f, const float* __restrict__ resid, float scale) {
  const int b = blockIdx.z;
  const int i0 = blockIdx.x * 64;
  const int n0 = blockIdx.y * 64;
  const int tid = threadIdx.x;
  const int wave = tid >> 5;
  const int lane = tid & 31;
  const int lh = lane & 15;
  const bool hig = lane >= 16;

  const bf16_t* Ab = A + (size_t)b * NSP * CCH;
  const bf16_t* arow = Ab + (size_t)(i0 + wave * 16 + lh) * CCH + (hig ? 8 : 0);
  const bf16_t* wrow[4];
#pragma unroll
  for (int t = 0; t < 4; ++t)
    wrow[t] = W + (size_t)(n0 + t * 16 + lh) * CCH + (hig ? 8 : 0);

  v8f acc[4] = {};
#pragma unroll 4
  for (int k0 = 0; k0 < CCH; k0 += 32) {
    v16bf af = bcat(*(const bf16x8*)(arow + k0), *(const bf16x8*)(arow + k0 + 16));
#pragma unroll
    for (int t = 0; t < 4; ++t) {
      v16bf bfg = bcat(*(const bf16x8*)(wrow[t] + k0),
                       *(const bf16x8*)(wrow[t] + k0 + 16));
      acc[t] = wmma_bf16(af, bfg, acc[t]);
    }
  }

  if constexpr (MODE == 0) {
    bf16_t* ob = out_bf + (size_t)b * NSP * CCH;
#pragma unroll
    for (int t = 0; t < 4; ++t) {
      const int c = n0 + t * 16 + lh;
      const float bv = bias[c];
#pragma unroll
      for (int r = 0; r < 8; ++r) {
        const int irow = i0 + wave * 16 + r + (hig ? 8 : 0);
        ob[(size_t)irow * CCH + c] = (bf16_t)((acc[t][r] + bv) * scale);
      }
    }
  } else {
    __shared__ float tile[64 * 65];
#pragma unroll
    for (int t = 0; t < 4; ++t)
#pragma unroll
      for (int r = 0; r < 8; ++r)
        tile[(t * 16 + lh) * 65 + wave * 16 + r + (hig ? 8 : 0)] = acc[t][r];
    __syncthreads();
    for (int e = tid; e < 64 * 64; e += 128) {
      const int cl = e >> 6, il = e & 63;
      const size_t idx = ((size_t)b * CCH + n0 + cl) * NSP + i0 + il;
      const float v = tile[cl * 65 + il] + bias[n0 + cl];
      if constexpr (MODE == 1)
        out_bf[idx] = (bf16_t)v;
      else
        out_f[idx] = resid[idx] + v;
    }
  }
}

// ---------------------------------------------------------------------------
// Flash attention. Block = 16 query rows, 4 waves.
//   qt [B][N][C] bf16 (scale folded), kt [B][N][C] bf16, vcj [B][C][N] bf16
//   ot [B][N][C] bf16  (O / softmax-denominator)
// Each wave owns O^T[16 rows][128 channels] = 8 f32 accumulator tiles.
// Per 64-key step: 16 WMMA for S + 16 WMMA for O update, online softmax.
// ---------------------------------------------------------------------------
__global__ __launch_bounds__(128) void flash_attn_kernel(
    const bf16_t* __restrict__ qt, const bf16_t* __restrict__ kt,
    const bf16_t* __restrict__ vcj, bf16_t* __restrict__ ot) {
  constexpr int QS = 520;  // 512 + 8 pad  -> conflict-free b128 reads
  constexpr int PS = 72;   // 64 + 8 pad
  __shared__ __align__(16) bf16_t q_lds[16 * QS];
  __shared__ __align__(16) bf16_t p_lds[16 * PS];
  __shared__ float red[128];

  const int b = blockIdx.y;
  const int i0 = blockIdx.x * 16;
  const int tid = threadIdx.x;
  const int wave = tid >> 5;
  const int lane = tid & 31;
  const int lh = lane & 15;
  const bool hig = lane >= 16;

  // stage Q tile (16 x 512 bf16) into LDS
  const bf16_t* qb = qt + ((size_t)b * NSP + i0) * CCH;
  for (int e = tid; e < 16 * CCH / 8; e += 128) {
    const int row = e >> 6;
    const int cc = (e & 63) * 8;
    *(bf16x8*)&q_lds[row * QS + cc] = *(const bf16x8*)&qb[(size_t)row * CCH + cc];
  }
  __syncthreads();

  const bf16_t* ql = q_lds + lh * QS + (hig ? 8 : 0);
  const bf16_t* pl = p_lds + lh * PS + (hig ? 8 : 0);
  const bf16_t* kb = kt + (size_t)b * NSP * CCH + (hig ? 8 : 0);
  const bf16_t* vb = vcj + (size_t)b * CCH * NSP + (hig ? 8 : 0);
  const bf16_t* vbt[8];
#pragma unroll
  for (int t = 0; t < 8; ++t)
    vbt[t] = vb + (size_t)(wave * 128 + t * 16 + lh) * NSP;

  float m_run[8], l_run[8];
  v8f o_acc[8] = {};
#pragma unroll
  for (int r = 0; r < 8; ++r) {
    m_run[r] = -1e30f;
    l_run[r] = 0.f;
  }

  for (int j0 = 0; j0 < NSP; j0 += 64) {
    // ---- S tile: this wave's 16 key columns, full 512 contraction ----
    const bf16_t* krow = kb + (size_t)(j0 + wave * 16 + lh) * CCH;
    v8f s = {};
#pragma unroll 4
    for (int c0 = 0; c0 < CCH; c0 += 32) {
      v16bf af = bcat(*(const bf16x8*)(ql + c0), *(const bf16x8*)(ql + c0 + 16));
      v16bf bfg = bcat(*(const bf16x8*)(krow + c0), *(const bf16x8*)(krow + c0 + 16));
      s = wmma_bf16(af, bfg, s);
    }

    __syncthreads();  // previous step's p_lds / red readers are done

    // ---- per-wave row max over 16 columns (cross-lane within half) ----
    float mw[8];
#pragma unroll
    for (int r = 0; r < 8; ++r) {
      float m = s[r];
      m = fmaxf(m, __shfl_xor(m, 1, 32));
      m = fmaxf(m, __shfl_xor(m, 2, 32));
      m = fmaxf(m, __shfl_xor(m, 4, 32));
      m = fmaxf(m, __shfl_xor(m, 8, 32));
      mw[r] = m;
    }
    if (lh == 0) {
#pragma unroll
      for (int r = 0; r < 8; ++r) red[wave * 16 + r + (hig ? 8 : 0)] = mw[r];
    }
    __syncthreads();

    float alpha[8];
#pragma unroll
    for (int r = 0; r < 8; ++r) {
      const int row = r + (hig ? 8 : 0);
      const float mstep = fmaxf(fmaxf(red[row], red[16 + row]),
                                fmaxf(red[32 + row], red[48 + row]));
      const float mnew = fmaxf(m_run[r], mstep);
      alpha[r] = __expf(m_run[r] - mnew);
      m_run[r] = mnew;
      const float pv = __expf(s[r] - mnew);
      p_lds[row * PS + wave * 16 + lh] = (bf16_t)pv;
      float ls = pv;
      ls += __shfl_xor(ls, 1, 32);
      ls += __shfl_xor(ls, 2, 32);
      ls += __shfl_xor(ls, 4, 32);
      ls += __shfl_xor(ls, 8, 32);
      if (lh == 0) red[64 + wave * 16 + row] = ls;
    }
    __syncthreads();

#pragma unroll
    for (int r = 0; r < 8; ++r) {
      const int row = r + (hig ? 8 : 0);
      l_run[r] = l_run[r] * alpha[r] + red[64 + row] + red[80 + row] +
                 red[96 + row] + red[112 + row];
    }
#pragma unroll
    for (int t = 0; t < 8; ++t)
#pragma unroll
      for (int r = 0; r < 8; ++r) o_acc[t][r] *= alpha[r];

    // ---- O += P * V^T over this wave's 128 channels ----
#pragma unroll
    for (int kk = 0; kk < 64; kk += 32) {
      v16bf pf = bcat(*(const bf16x8*)(pl + kk), *(const bf16x8*)(pl + kk + 16));
#pragma unroll
      for (int t = 0; t < 8; ++t) {
        v16bf vf = bcat(*(const bf16x8*)(vbt[t] + j0 + kk),
                        *(const bf16x8*)(vbt[t] + j0 + kk + 16));
        o_acc[t] = wmma_bf16(pf, vf, o_acc[t]);
      }
    }
  }

  // epilogue: normalize and store O^T as bf16 [N][C]
  bf16_t* ob = ot + ((size_t)b * NSP + i0) * CCH;
#pragma unroll
  for (int r = 0; r < 8; ++r) {
    const float inv = 1.0f / l_run[r];
    const int irow = r + (hig ? 8 : 0);
#pragma unroll
    for (int t = 0; t < 8; ++t) {
      ob[(size_t)irow * CCH + wave * 128 + t * 16 + lh] =
          (bf16_t)(o_acc[t][r] * inv);
    }
  }
}

// ---------------------------------------------------------------------------
extern "C" void kernel_launch(void* const* d_in, const int* in_sizes, int n_in,
                              void* d_out, int out_size, void* d_ws, size_t ws_size,
                              hipStream_t stream) {
  const float* x     = (const float*)d_in[0];
  const float* gamma = (const float*)d_in[1];
  const float* beta  = (const float*)d_in[2];
  const float* wq = (const float*)d_in[3];
  const float* bq = (const float*)d_in[4];
  const float* wk = (const float*)d_in[5];
  const float* bk = (const float*)d_in[6];
  const float* wv = (const float*)d_in[7];
  const float* bv = (const float*)d_in[8];
  const float* wo = (const float*)d_in[9];
  const float* bo = (const float*)d_in[10];
  float* out = (float*)d_out;

  char* w = (char*)d_ws;
  auto carve = [&](size_t bytes) -> char* {
    char* p = w;
    w += (bytes + 255) & ~(size_t)255;
    return p;
  };
  const size_t bnc = (size_t)BATCH * NSP * CCH;
  float*  mean = (float*)carve(BATCH * NGRP * sizeof(float));
  float*  rstd = (float*)carve(BATCH * NGRP * sizeof(float));
  bf16_t* nxt  = (bf16_t*)carve(bnc * 2);
  bf16_t* qtb  = (bf16_t*)carve(bnc * 2);
  bf16_t* ktb  = (bf16_t*)carve(bnc * 2);
  bf16_t* vcj  = (bf16_t*)carve(bnc * 2);
  bf16_t* otb  = (bf16_t*)carve(bnc * 2);
  bf16_t* wqb  = (bf16_t*)carve((size_t)CCH * CCH * 2);
  bf16_t* wkb  = (bf16_t*)carve((size_t)CCH * CCH * 2);
  bf16_t* wvb  = (bf16_t*)carve((size_t)CCH * CCH * 2);
  bf16_t* wob  = (bf16_t*)carve((size_t)CCH * CCH * 2);

  gn_stats_kernel<<<BATCH * NGRP, 256, 0, stream>>>(x, mean, rstd);
  gn_apply_kernel<<<dim3(NSP / 32, CCH / 32, BATCH), dim3(32, 8), 0, stream>>>(
      x, gamma, beta, mean, rstd, nxt);

  const int wn = CCH * CCH;
  cvt_bf16_kernel<<<(wn + 255) / 256, 256, 0, stream>>>(wq, wqb, wn);
  cvt_bf16_kernel<<<(wn + 255) / 256, 256, 0, stream>>>(wk, wkb, wn);
  cvt_bf16_kernel<<<(wn + 255) / 256, 256, 0, stream>>>(wv, wvb, wn);
  cvt_bf16_kernel<<<(wn + 255) / 256, 256, 0, stream>>>(wo, wob, wn);

  const dim3 ggrid(NSP / 64, CCH / 64, BATCH);
  const float qscale = 0.04419417382415922f;  // 1/sqrt(512)
  gemm_nc_kernel<0><<<ggrid, 128, 0, stream>>>(nxt, wqb, bq, qtb, nullptr, nullptr, qscale);
  gemm_nc_kernel<0><<<ggrid, 128, 0, stream>>>(nxt, wkb, bk, ktb, nullptr, nullptr, 1.0f);
  gemm_nc_kernel<1><<<ggrid, 128, 0, stream>>>(nxt, wvb, bv, vcj, nullptr, nullptr, 1.0f);

  flash_attn_kernel<<<dim3(NSP / 16, BATCH), 128, 0, stream>>>(qtb, ktb, vcj, otb);

  gemm_nc_kernel<2><<<ggrid, 128, 0, stream>>>(otb, wob, bo, nullptr, out, x, 1.0f);
}
